// GCN_t2_73658689126525
// MI455X (gfx1250) — matrix operand order repaired and checked by reading, not compiled, not run on hardware
//
#include <hip/hip_runtime.h>
#include <hip/hip_bf16.h>
#include <math.h>

typedef __attribute__((ext_vector_type(16))) __bf16 v16bf;
typedef __attribute__((ext_vector_type(8)))  __bf16 v8bf;
typedef __attribute__((ext_vector_type(8)))  float  v8f;

#define KS 4  // spline kernel size

__device__ __forceinline__ unsigned short f2bf(float f) {
  union { float f; unsigned u; } x; x.f = f;
  unsigned u = x.u;
  u += 0x7FFFu + ((u >> 16) & 1u);   // round-to-nearest-even
  return (unsigned short)(u >> 16);
}

__device__ __forceinline__ void atomAddF(float* p, float v) {
#if defined(__HIP_DEVICE_COMPILE__)
  unsafeAtomicAdd(p, v);             // lowers to global_atomic_add_f32
#else
  *p += v;
#endif
}

// ---- convert x (f32 [N,Fin]) to zero-padded bf16 [N,Kpad] -------------------
__global__ void k_pack_xbf(const float* __restrict__ x, unsigned short* __restrict__ xbf,
                           int N, int Fin, int Kpad) {
  long long i = (long long)blockIdx.x * blockDim.x + threadIdx.x;
  long long total = (long long)N * Kpad;
  if (i >= total) return;
  int n = (int)(i / Kpad), k = (int)(i % Kpad);
  xbf[i] = (k < Fin) ? f2bf(x[(long long)n * Fin + k]) : (unsigned short)0;
}

// ---- pack [W | root] into bf16 column-major B: Bcm[col*Kpad + k] ------------
// cols [0, KS*Fout): col = kk*Fout + o -> W[kk][k][o]
// cols [KS*Fout, (KS+1)*Fout): root[k][col - KS*Fout];  rest zero
__global__ void k_pack_b(const float* __restrict__ W, const float* __restrict__ root,
                         unsigned short* __restrict__ Bcm,
                         int Fin, int Kpad, int Fout, int Ncols) {
  int i = blockIdx.x * blockDim.x + threadIdx.x;
  int total = Ncols * Kpad;
  if (i >= total) return;
  int col = i / Kpad, k = i % Kpad;
  float v = 0.0f;
  if (k < Fin) {
    if (col < KS * Fout) {
      int kk = col / Fout, o = col % Fout;
      v = W[((long long)kk * Fin + k) * Fout + o];
    } else if (col < (KS + 1) * Fout) {
      v = root[(long long)k * Fout + (col - KS * Fout)];
    }
  }
  Bcm[(long long)col * Kpad + k] = f2bf(v);
}

// ---- bf16 WMMA GEMM: C[M,ldc] = A[M,lda](bf16) * Bcm(col-major, K contig) ---
// one wave per 16x16 output tile, K-loop over 32-chunks
__global__ void k_gemm_wmma_bf16(const unsigned short* __restrict__ Abf,
                                 const unsigned short* __restrict__ Bbf,
                                 float* __restrict__ C,
                                 int M, int Ncols, int Ksteps, int lda, int ldbk, int ldc) {
  int wave = blockIdx.x * (blockDim.x >> 5) + (threadIdx.x >> 5);
  int ntn = Ncols >> 4;
  int total = (M >> 4) * ntn;
  if (wave >= total) return;              // wave-uniform: EXEC stays all-1s
  int mt = wave / ntn, nt = wave % ntn;
  int lane = threadIdx.x & 31;
  int l  = lane & 15;
  int hi = lane >> 4;

  const __bf16* A = reinterpret_cast<const __bf16*>(Abf);
  const __bf16* B = reinterpret_cast<const __bf16*>(Bbf);

  int row = (mt << 4) + l;                // A rows / D cols use lanes 0-15 twice
  int col = (nt << 4) + l;

  v8f c = {0.f, 0.f, 0.f, 0.f, 0.f, 0.f, 0.f, 0.f};
  for (int ks = 0; ks < Ksteps; ++ks) {
    int kb = ks << 5;
    // A frag (16x32 bf16): lane half 0 -> K kb+0..7 & kb+16..23 ; half 1 -> +8 / +24
    const __bf16* ap = A + (long long)row * lda + kb + (hi ? 8 : 0);
    v8bf alo = *reinterpret_cast<const v8bf*>(ap);
    v8bf ahi = *reinterpret_cast<const v8bf*>(ap + 16);
    v16bf a = __builtin_shufflevector(alo, ahi, 0,1,2,3,4,5,6,7,8,9,10,11,12,13,14,15);
    // B frag (32x16 bf16): lane half 0 -> K kb+0..7 & kb+16..23 ; half 1 -> +8 / +24
    const __bf16* bp = B + (long long)col * ldbk + kb + (hi ? 8 : 0);
    v8bf blo = *reinterpret_cast<const v8bf*>(bp);
    v8bf bhi = *reinterpret_cast<const v8bf*>(bp + 16);
    v16bf b = __builtin_shufflevector(blo, bhi, 0,1,2,3,4,5,6,7,8,9,10,11,12,13,14,15);
    c = __builtin_amdgcn_wmma_f32_16x16x32_bf16(false, a, false, b, (short)0, c, false, false);
  }
  // D layout: VGPR r, lanes 0-15 -> M=r, lanes 16-31 -> M=r+8, N=lane&15
  float* crow = C + (long long)((mt << 4) + (hi ? 8 : 0)) * ldc + (nt << 4) + l;
#pragma unroll
  for (int r = 0; r < 8; ++r) crow[(long long)r * ldc] = c[r];
}

// ---- per-edge degree --------------------------------------------------------
__global__ void k_deg(const long long* __restrict__ ei, float* __restrict__ deg, long long E) {
  long long e = (long long)blockIdx.x * blockDim.x + threadIdx.x;
  if (e >= E) return;
  atomAddF(deg + (int)ei[E + e], 1.0f);
}

__device__ __forceinline__ void spline_w(float u, int& k0, int& k1, float& w0, float& w1) {
  float v = u * (float)(KS - 1);
  float vf = floorf(v);
  float fr = v - vf;
  int k = (int)vf;
  k0 = k < 0 ? 0 : (k > KS - 1 ? KS - 1 : k);
  k1 = (k0 + 1 > KS - 1) ? KS - 1 : k0 + 1;
  w0 = 1.0f - fr; w1 = fr;
}

// ---- edge scatter, Fout = 32, float4-vectorized -----------------------------
__global__ void k_edge32(const long long* __restrict__ ei, const float* __restrict__ ea,
                         const float* __restrict__ XW, float* __restrict__ agg,
                         long long E, int ld) {
  long long e = (long long)blockIdx.x * blockDim.x + threadIdx.x;
  if (e >= E) return;
  int s = (int)ei[e], d = (int)ei[E + e];
  int k0, k1; float w0, w1;
  spline_w(ea[e], k0, k1, w0, w1);
  const float4* r0 = reinterpret_cast<const float4*>(XW + (long long)s * ld + k0 * 32);
  const float4* r1 = reinterpret_cast<const float4*>(XW + (long long)s * ld + k1 * 32);
  float* ag = agg + (long long)d * 32;
#pragma unroll
  for (int j = 0; j < 8; ++j) {
    float4 a = r0[j], b = r1[j];
    atomAddF(ag + 4 * j + 0, w0 * a.x + w1 * b.x);
    atomAddF(ag + 4 * j + 1, w0 * a.y + w1 * b.y);
    atomAddF(ag + 4 * j + 2, w0 * a.z + w1 * b.z);
    atomAddF(ag + 4 * j + 3, w0 * a.w + w1 * b.w);
  }
}

// ---- edge scatter, Fout = 10, scalar ----------------------------------------
__global__ void k_edge10(const long long* __restrict__ ei, const float* __restrict__ ea,
                         const float* __restrict__ XW, float* __restrict__ agg,
                         long long E, int ld) {
  long long e = (long long)blockIdx.x * blockDim.x + threadIdx.x;
  if (e >= E) return;
  int s = (int)ei[e], d = (int)ei[E + e];
  int k0, k1; float w0, w1;
  spline_w(ea[e], k0, k1, w0, w1);
  const float* r0 = XW + (long long)s * ld + k0 * 10;
  const float* r1 = XW + (long long)s * ld + k1 * 10;
  float* ag = agg + (long long)d * 10;
#pragma unroll
  for (int o = 0; o < 10; ++o) atomAddF(ag + o, w0 * r0[o] + w1 * r1[o]);
}

// ---- finalize layer 1: mean + root + bias, ELU, emit bf16 -------------------
__global__ void k_fin1(const float* __restrict__ agg, const float* __restrict__ deg,
                       const float* __restrict__ XWR, const float* __restrict__ bias,
                       unsigned short* __restrict__ hbf, int N) {
  long long i = (long long)blockIdx.x * blockDim.x + threadIdx.x;
  if (i >= (long long)N * 32) return;
  int n = (int)(i >> 5), o = (int)(i & 31);
  float dg = fmaxf(deg[n], 1.0f);
  float v = agg[i] / dg + XWR[(long long)n * 160 + 128 + o] + bias[o];
  v = v > 0.0f ? v : (expf(v) - 1.0f);  // ELU
  hbf[i] = f2bf(v);
}

// ---- finalize layer 2: mean + root + bias, log_softmax ----------------------
__global__ void k_fin2(const float* __restrict__ agg2, const float* __restrict__ deg,
                       const float* __restrict__ XWR2, const float* __restrict__ bias2,
                       float* __restrict__ out, int N) {
  int n = blockIdx.x * blockDim.x + threadIdx.x;
  if (n >= N) return;
  float dg = fmaxf(deg[n], 1.0f);
  float v[10], m = -1e30f;
#pragma unroll
  for (int o = 0; o < 10; ++o) {
    v[o] = agg2[(long long)n * 10 + o] / dg + XWR2[(long long)n * 64 + 40 + o] + bias2[o];
    m = fmaxf(m, v[o]);
  }
  float ssum = 0.0f;
#pragma unroll
  for (int o = 0; o < 10; ++o) ssum += expf(v[o] - m);
  float ls = logf(ssum);
#pragma unroll
  for (int o = 0; o < 10; ++o) out[(long long)n * 10 + o] = v[o] - m - ls;
}

extern "C" void kernel_launch(void* const* d_in, const int* in_sizes, int n_in,
                              void* d_out, int out_size, void* d_ws, size_t ws_size,
                              hipStream_t stream) {
  (void)n_in; (void)out_size; (void)ws_size;
  const float*     x     = (const float*)d_in[0];
  const long long* ei    = (const long long*)d_in[1];   // int64 [2,E]
  const float*     ea    = (const float*)d_in[2];
  const float*     W1    = (const float*)d_in[3];
  const float*     root1 = (const float*)d_in[4];
  const float*     bias1 = (const float*)d_in[5];
  const float*     W2    = (const float*)d_in[6];
  const float*     root2 = (const float*)d_in[7];
  const float*     bias2 = (const float*)d_in[8];

  const int FIN = 48, HID = 32, CLS = 10;
  const int KPAD1 = 64, NC1 = (KS + 1) * HID;   // 160 = 4*32 spline + 32 root
  const int KPAD2 = 32, NC2 = 64;               // 40 spline + 10 root + 14 pad
  const int N = in_sizes[0] / FIN;
  const long long E = (long long)in_sizes[2];

  char* ws = (char*)d_ws;
  size_t off = 0;
  auto alloc = [&](size_t bytes) -> void* {
    void* p = ws + off; off = (off + bytes + 255) & ~(size_t)255; return p;
  };
  unsigned short* xbf  = (unsigned short*)alloc((size_t)N * KPAD1 * 2);
  unsigned short* B1   = (unsigned short*)alloc((size_t)NC1 * KPAD1 * 2);
  float*          XWR1 = (float*)alloc((size_t)N * NC1 * 4);
  float*          deg  = (float*)alloc((size_t)N * 4);
  float*          agg1 = (float*)alloc((size_t)N * HID * 4);
  unsigned short* hbf  = (unsigned short*)alloc((size_t)N * KPAD2 * 2);
  unsigned short* B2   = (unsigned short*)alloc((size_t)NC2 * KPAD2 * 2);
  float*          XWR2 = (float*)alloc((size_t)N * NC2 * 4);
  float*          agg2 = (float*)alloc((size_t)N * CLS * 4);

  hipMemsetAsync(deg,  0, (size_t)N * 4,       stream);
  hipMemsetAsync(agg1, 0, (size_t)N * HID * 4, stream);
  hipMemsetAsync(agg2, 0, (size_t)N * CLS * 4, stream);

  // ---- layer 1 ----
  { long long t = (long long)N * KPAD1;
    k_pack_xbf<<<(unsigned)((t + 255) / 256), 256, 0, stream>>>(x, xbf, N, FIN, KPAD1); }
  { int t = NC1 * KPAD1;
    k_pack_b<<<(t + 255) / 256, 256, 0, stream>>>(W1, root1, B1, FIN, KPAD1, HID, NC1); }
  { int tiles = (N / 16) * (NC1 / 16);
    k_gemm_wmma_bf16<<<(tiles + 7) / 8, 256, 0, stream>>>(
        xbf, B1, XWR1, N, NC1, KPAD1 / 32, KPAD1, KPAD1, NC1); }
  k_deg<<<(unsigned)((E + 255) / 256), 256, 0, stream>>>(ei, deg, E);
  k_edge32<<<(unsigned)((E + 255) / 256), 256, 0, stream>>>(ei, ea, XWR1, agg1, E, NC1);
  { long long t = (long long)N * HID;
    k_fin1<<<(unsigned)((t + 255) / 256), 256, 0, stream>>>(agg1, deg, XWR1, bias1, hbf, N); }

  // ---- layer 2 ----
  { int t = NC2 * KPAD2;
    k_pack_b<<<(t + 255) / 256, 256, 0, stream>>>(W2, root2, B2, HID, KPAD2, CLS, NC2); }
  { int tiles = (N / 16) * (NC2 / 16);
    k_gemm_wmma_bf16<<<(tiles + 7) / 8, 256, 0, stream>>>(
        hbf, B2, XWR2, N, NC2, KPAD2 / 32, KPAD2, KPAD2, NC2); }
  k_edge10<<<(unsigned)((E + 255) / 256), 256, 0, stream>>>(ei, ea, XWR2, agg2, E, NC2);
  k_fin2<<<(N + 255) / 256, 256, 0, stream>>>(agg2, deg, XWR2, bias2, (float*)d_out, N);
}
